// GIN_61976378081728
// MI455X (gfx1250) — matrix operand order repaired and checked by reading, not compiled, not run on hardware
//
#include <hip/hip_runtime.h>

// ---------------------------------------------------------------------------
// GINConv: out = MLP((1+eps)*x + segment_sum(x[src], dst))
//   MLP: Linear(1,20) -> ReLU -> Linear(20,20) -> ReLU -> Linear(20,1)
//
// Phase 1 (dominant): stream 512MB of int64 edge indices (NT loads), gather
//   x[src] (L2-resident, 4MB), atomic-add into agg[dst] (L2-resident).
// Phase 2: per-wave 16-node batch; middle 20x20 layer via two
//   v_wmma_f32_16x16x32_f16 (K=20 padded to 32, N=20 -> two 16-col tiles).
// ---------------------------------------------------------------------------

typedef __attribute__((ext_vector_type(16))) _Float16 v16h;
typedef __attribute__((ext_vector_type(8)))  float    v8f;
typedef __attribute__((ext_vector_type(2)))  long long ll2;

// ---- Phase 0: agg[i] = (1+eps) * x[i] -------------------------------------
__global__ void gin_init_kernel(const float* __restrict__ x,
                                const float* __restrict__ eps,
                                float* __restrict__ agg, int n) {
    int i = blockIdx.x * blockDim.x + threadIdx.x;
    if (i < n) agg[i] = (1.0f + eps[0]) * x[i];
}

// ---- Phase 1: agg[dst[e]] += x[src[e]] ------------------------------------
// 2 edges per thread: two 16B non-temporal loads (edge stream is one-shot,
// keep it out of the way of the hot 4MB x/agg working set in L2).
__global__ void gin_scatter_kernel(const float* __restrict__ x,
                                   const long long* __restrict__ src,
                                   const long long* __restrict__ dst,
                                   float* __restrict__ agg,
                                   long long n_edges) {
    long long e = ((long long)blockIdx.x * blockDim.x + threadIdx.x) * 2;
    if (e + 1 < n_edges) {
        ll2 s = __builtin_nontemporal_load((const ll2*)(src + e));
        ll2 d = __builtin_nontemporal_load((const ll2*)(dst + e));
        atomicAdd(&agg[(int)d.x], x[(int)s.x]);
        atomicAdd(&agg[(int)d.y], x[(int)s.y]);
    } else if (e < n_edges) {
        long long s = src[e], d = dst[e];
        atomicAdd(&agg[(int)d], x[(int)s]);
    }
}

// ---- Phase 2: MLP, 16 nodes per wave, WMMA for the 20x20 layer ------------
__global__ void gin_mlp_kernel(const float* __restrict__ agg,
                               const float* __restrict__ w1,
                               const float* __restrict__ b1,
                               const float* __restrict__ w2,   // [20][20] row-major (k,n)
                               const float* __restrict__ b2,
                               const float* __restrict__ w3,
                               const float* __restrict__ b3,
                               float* __restrict__ out, int n) {
    const int lane = threadIdx.x & 31;
    const int wave = threadIdx.x >> 5;
    const int half = lane >> 4;       // K-half / M-half selector (ISA layout)
    const int m    = lane & 15;       // node row (A) / column N (B,C,D)

    const long long node_base = (long long)blockIdx.x * 128 + (long long)wave * 16;

    // h0 for node (node_base + m); clamp OOB to 0 (outputs are guarded).
    long long nm = node_base + m;
    float h0 = (nm < n) ? agg[nm] : 0.0f;

    // ---- Layer 1 + pack A (16x32 f16, K=20 zero-padded) ----
    // A element j -> VGPR j/2, 16-bit slot j&1:
    //   k = (j/2 < 4 ? 0 : 16) + 8*half + 2*((j/2)&3) + (j&1)
    v16h a;
#pragma unroll
    for (int j = 0; j < 16; ++j) {
        int v = j >> 1, p = j & 1;
        int k = ((v < 4) ? 0 : 16) + half * 8 + 2 * (v & 3) + p;
        float h1 = 0.0f;
        if (k < 20) h1 = fmaxf(fmaf(h0, w1[k], b1[k]), 0.0f);
        a[j] = (_Float16)h1;
    }

    // ---- Pack B (32x16 f16) for two N-tiles: n0 = m, n1 = 16+m ----
    // B element j -> k = (j < 8 ? 8*half + j : 16 + 8*half + (j-8)), column = n
    v16h bt0, bt1;
#pragma unroll
    for (int j = 0; j < 16; ++j) {
        int k = (j < 8) ? (half * 8 + j) : (16 + half * 8 + (j - 8));
        float wa = 0.0f, wb = 0.0f;
        if (k < 20) {
            wa = w2[k * 20 + m];                      // n0 = m < 16 < 20
            if (m < 4) wb = w2[k * 20 + 16 + m];      // n1 = 16+m, valid if < 20
        }
        bt0[j] = (_Float16)wa;
        bt1[j] = (_Float16)wb;
    }

    // ---- C init = broadcast b2[N] (C/D element r: M = r + 8*half, N = m) ----
    float bc0 = b2[m];
    float bc1 = (m < 4) ? b2[16 + m] : 0.0f;
    v8f acc0, acc1;
#pragma unroll
    for (int r = 0; r < 8; ++r) { acc0[r] = bc0; acc1[r] = bc1; }

    // D = A x B + C   (f16 in, f32 accumulate)
    acc0 = __builtin_amdgcn_wmma_f32_16x16x32_f16(false, a, false, bt0,
                                                  (short)0, acc0, false, false);
    acc1 = __builtin_amdgcn_wmma_f32_16x16x32_f16(false, a, false, bt1,
                                                  (short)0, acc1, false, false);

    // ---- ReLU + Layer 3 dot(H2[node], w3) + b3 ----
    float w3a = w3[m];                          // N = m
    float w3b = (m < 4) ? w3[16 + m] : 0.0f;    // N = 16+m
    float bias3 = b3[0];
#pragma unroll
    for (int r = 0; r < 8; ++r) {
        float t = fmaxf(acc0[r], 0.0f) * w3a + fmaxf(acc1[r], 0.0f) * w3b;
        // reduce across the 16 lanes of this half-group (masks < 16)
        t += __shfl_xor(t, 1, 32);
        t += __shfl_xor(t, 2, 32);
        t += __shfl_xor(t, 4, 32);
        t += __shfl_xor(t, 8, 32);
        long long node = node_base + half * 8 + r;
        if (m == 0 && node < n) out[node] = t + bias3;
    }
}

// ---------------------------------------------------------------------------
extern "C" void kernel_launch(void* const* d_in, const int* in_sizes, int n_in,
                              void* d_out, int out_size, void* d_ws, size_t ws_size,
                              hipStream_t stream) {
    const float*     x   = (const float*)d_in[0];
    const long long* ei  = (const long long*)d_in[1];   // int64 [2, E]
    const float*     eps = (const float*)d_in[2];
    const float*     w1  = (const float*)d_in[3];
    const float*     b1  = (const float*)d_in[4];
    const float*     w2  = (const float*)d_in[5];
    const float*     b2  = (const float*)d_in[6];
    const float*     w3  = (const float*)d_in[7];
    const float*     b3  = (const float*)d_in[8];

    float* out = (float*)d_out;
    float* agg = (float*)d_ws;                 // N floats = 4MB scratch

    const int       n_nodes = in_sizes[0];                 // x is [N,1]
    const long long n_edges = (long long)in_sizes[1] / 2;  // edge_index is [2,E]
    const long long* src = ei;
    const long long* dst = ei + n_edges;

    // Phase 0: agg = (1+eps)*x
    gin_init_kernel<<<(n_nodes + 255) / 256, 256, 0, stream>>>(x, eps, agg, n_nodes);

    // Phase 1: scatter-add (2 edges / thread)
    long long pairs = (n_edges + 1) / 2;
    int sblocks = (int)((pairs + 255) / 256);
    gin_scatter_kernel<<<sblocks, 256, 0, stream>>>(x, src, dst, agg, n_edges);

    // Phase 2: MLP with WMMA, 128 nodes per 256-thread block (8 waves x 16)
    gin_mlp_kernel<<<(n_nodes + 127) / 128, 256, 0, stream>>>(
        agg, w1, b1, w2, b2, w3, b3, out, n_nodes);
}
